// SelfAttention_90640989815518
// MI455X (gfx1250) — matrix operand order repaired and checked by reading, not compiled, not run on hardware
//
#include <hip/hip_runtime.h>
#include <hip/hip_bf16.h>

// MI455X (gfx1250, wave32) fused self-attention block.
// Compute-bound on v_wmma_f32_16x16x32_f16 (~69 GFLOP attention core).
// Flash attention avoids the 268MB score matrix; K/V blocks are staged into
// LDS via double-buffered GLOBAL_LOAD_ASYNC_TO_LDS_B128 (ASYNCcnt) so each
// workgroup reads K/V once (8x L2-traffic cut vs per-wave streaming).

typedef __attribute__((ext_vector_type(16))) _Float16 v16h;
typedef __attribute__((ext_vector_type(8)))  _Float16 v8h;
typedef __attribute__((ext_vector_type(8)))  float    v8f;
typedef int v4i_ __attribute__((vector_size(16)));   // matches builtin param

#define B_  4
#define C_  256
#define D_  256
#define N_  4096
#define NG  32
#define CPG 8
#define GN_EPS 1e-5f

// ---- CDNA5 async global->LDS copy (ASYNCcnt) with safe fallback ----------
#if __has_builtin(__builtin_amdgcn_global_load_async_to_lds_b128)
#define HAVE_ASYNC 1
#define GLOAD_LDS128(g, l)                                                   \
  __builtin_amdgcn_global_load_async_to_lds_b128(                            \
      (__attribute__((address_space(1))) v4i_*)(g),                          \
      (__attribute__((address_space(3))) v4i_*)(l), 0, 0)
#if __has_builtin(__builtin_amdgcn_s_wait_asynccnt)
#define WAIT_ASYNC(n) __builtin_amdgcn_s_wait_asynccnt(n)
#else
#define WAIT_ASYNC(n) asm volatile("s_wait_asynccnt %0" ::"i"(n) : "memory")
#endif
#else
#define HAVE_ASYNC 0
#define GLOAD_LDS128(g, l) (*(v8h*)(l) = *(const v8h*)(g))
#define WAIT_ASYNC(n)
#endif

__device__ __forceinline__ v8f wmma_f16(v16h a, v16h b, v8f c) {
  return __builtin_amdgcn_wmma_f32_16x16x32_f16(false, a, false, b,
                                                (short)0, c, false, false);
}

// A operand (16x32 f16): lane = row M; lo half K {0..7,16..23}, hi half
// K {8..15,24..31}. rowp = K-contiguous row start.
__device__ __forceinline__ v16h load_a16(const _Float16* rowp, int half_) {
  v8h c0 = *(const v8h*)(rowp + half_ * 8);
  v8h c1 = *(const v8h*)(rowp + 16 + half_ * 8);
  v16h r;
#pragma unroll
  for (int i = 0; i < 8; ++i) { r[i] = c0[i]; r[i + 8] = c1[i]; }
  return r;
}

// B operand (32x16 f16): lane = column N; lo half K {0..7,16..23}, hi half
// K {8..15,24..31}.
__device__ __forceinline__ v16h load_b16(const _Float16* colp, int half_) {
  v8h c0 = *(const v8h*)(colp + half_ * 8);
  v8h c1 = *(const v8h*)(colp + 16 + half_ * 8);
  v16h r;
#pragma unroll
  for (int i = 0; i < 8; ++i) { r[i] = c0[i]; r[i + 8] = c1[i]; }
  return r;
}

// ---------------------------------------------------------------------------
// Kernel 0a: one-time weight conversion f32 -> f16 (same [out][in] layout).
__global__ void __launch_bounds__(256) wcvt_kernel(
    const float* __restrict__ w, _Float16* __restrict__ wh, int n) {
  const int i = blockIdx.x * 256 + threadIdx.x;
  if (i < n) wh[i] = (_Float16)w[i];
}

// Kernel 0b: x [B,C,N] f32 -> Xh [B,N,C] f16 via LDS 32x32 tile transpose.
__global__ void __launch_bounds__(256) xpose_kernel(
    const float* __restrict__ x, _Float16* __restrict__ Xh) {
  __shared__ float tile[32][33];
  const int bc = blockIdx.x;              // over B*C/32
  const int b  = bc / (C_ / 32);
  const int c0 = (bc % (C_ / 32)) * 32;
  const int n0 = blockIdx.y * 32;
  const int tx = threadIdx.x & 31, ty = threadIdx.x >> 5;
#pragma unroll
  for (int j = 0; j < 4; ++j) {
    const int c = c0 + ty + j * 8;
    tile[ty + j * 8][tx] = x[((size_t)b * C_ + c) * N_ + n0 + tx];
  }
  __syncthreads();
#pragma unroll
  for (int j = 0; j < 4; ++j) {
    const int n = n0 + ty + j * 8;
    Xh[((size_t)b * N_ + n) * C_ + c0 + tx] = (_Float16)tile[tx][ty + j * 8];
  }
}

// ---------------------------------------------------------------------------
// Kernel 1: QKV projection, pure f16 WMMA. out[n,d] = sum_c Xh[n,c]*W[d,c]+b.
// Q scaled by 1/sqrt(D)=1/16. Q,K -> [B,N,D] f16; V -> [B,D,N] f16.
__global__ void __launch_bounds__(256) qkv_proj_kernel(
    const _Float16* __restrict__ Xh,
    const _Float16* __restrict__ Wqh, const float* __restrict__ bq,
    const _Float16* __restrict__ Wkh, const float* __restrict__ bk,
    const _Float16* __restrict__ Wvh, const float* __restrict__ bv,
    _Float16* __restrict__ Qh, _Float16* __restrict__ Kh,
    _Float16* __restrict__ Vh) {
  const int wave  = (blockIdx.x * blockDim.x + threadIdx.x) >> 5;
  const int ln    = threadIdx.x & 31;
  const int half_ = ln >> 4;
  const int lid   = ln & 15;
  const int b     = wave / (N_ / 16);
  const int q0    = (wave % (N_ / 16)) * 16;

  v16h xa[8];
  {
    const _Float16* rowp = Xh + ((size_t)b * N_ + q0 + lid) * C_;
#pragma unroll
    for (int kb = 0; kb < 8; ++kb) xa[kb] = load_a16(rowp + kb * 32, half_);
  }

  for (int m = 0; m < 3; ++m) {
    const _Float16* W    = (m == 0) ? Wqh : ((m == 1) ? Wkh : Wvh);
    const float*    bias = (m == 0) ? bq : ((m == 1) ? bk : bv);
    const float     scl  = (m == 0) ? 0.0625f : 1.0f;
    for (int dt = 0; dt < 16; ++dt) {
      const int d = dt * 16 + lid;
      const _Float16* wrow = W + (size_t)d * C_;
      v8f acc = {};
#pragma unroll
      for (int kb = 0; kb < 8; ++kb)
        acc = wmma_f16(xa[kb], load_b16(wrow + kb * 32, half_), acc);
      const float bd = bias[d];
      if (m < 2) {
        _Float16* dst = (m == 0) ? Qh : Kh;
#pragma unroll
        for (int r = 0; r < 8; ++r) {
          const int row = q0 + r + half_ * 8;
          dst[((size_t)b * N_ + row) * D_ + d] = (_Float16)((acc[r] + bd) * scl);
        }
      } else {
#pragma unroll
        for (int r = 0; r < 8; ++r) {
          const int row = q0 + r + half_ * 8;
          Vh[((size_t)b * D_ + d) * N_ + row] = (_Float16)(acc[r] + bd);
        }
      }
    }
  }
}

// ---------------------------------------------------------------------------
// Kernel 2: flash attention. 8 waves/WG share async-staged K/V blocks in LDS;
// each wave owns a 16-query tile with a 128-VGPR f32 output accumulator.
#define KST_H (2 * 32 * 256)   // halves in K stage (double buffered)
#define VST_H (2 * 256 * 32)
__global__ void __launch_bounds__(256) flash_attn_kernel(
    const _Float16* __restrict__ Qh, const _Float16* __restrict__ Kh,
    const _Float16* __restrict__ Vh, _Float16* __restrict__ AO) {
  extern __shared__ _Float16 smem[];
  _Float16* KstB = smem;                  // [2][32 keys][256 d]
  _Float16* VstB = smem + KST_H;          // [2][256 d][32 m]
  _Float16* pb   = smem + KST_H + VST_H;  // [8 waves][16][32]

  const int tid   = threadIdx.x;
  const int wv    = tid >> 5;
  const int ln    = tid & 31;
  const int half_ = ln >> 4;
  const int lid   = ln & 15;
  const int wave  = blockIdx.x * 8 + wv;
  const int b     = wave / (N_ / 16);
  const int q0    = (wave % (N_ / 16)) * 16;

  const _Float16* Qb = Qh + (size_t)b * N_ * D_;
  const _Float16* Kb = Kh + (size_t)b * N_ * D_;
  const _Float16* Vb = Vh + (size_t)b * D_ * N_;

  v16h qa[8];
  {
    const _Float16* rowp = Qb + (size_t)(q0 + lid) * D_;
#pragma unroll
    for (int kb = 0; kb < 8; ++kb) qa[kb] = load_a16(rowp + kb * 32, half_);
  }

  v8f out[16];
#pragma unroll
  for (int t = 0; t < 16; ++t) out[t] = (v8f){};
  float mrun[8], lrun[8];
#pragma unroll
  for (int r = 0; r < 8; ++r) { mrun[r] = -1e30f; lrun[r] = 0.0f; }

  // Cooperative stage of one 32-key block (K: 16KB contiguous; V: 256 rows
  // of 64B) -- 8 async b128 copies per thread.
  auto stage = [&](int buf, int m0) {
    _Float16* kd = KstB + buf * (32 * 256);
    _Float16* vd = VstB + buf * (256 * 32);
    const _Float16* kg = Kb + (size_t)m0 * D_;
#pragma unroll
    for (int j = 0; j < 4; ++j) {
      const int ci = tid + 256 * j;       // chunk of 8 halves
      GLOAD_LDS128(kg + ci * 8, kd + ci * 8);
    }
#pragma unroll
    for (int j = 0; j < 4; ++j) {
      const int ci = tid + 256 * j;
      const int d = ci >> 2, mo = (ci & 3) * 8;
      GLOAD_LDS128(Vb + (size_t)d * N_ + m0 + mo, vd + d * 32 + mo);
    }
  };

  stage(0, 0);
  for (int blk = 0; blk < N_ / 32; ++blk) {
    const int m0  = blk * 32;
    const int cur = blk & 1;
    if (blk + 1 < N_ / 32) {
      stage(cur ^ 1, m0 + 32);
      WAIT_ASYNC(8);                       // current buffer's copies done
    } else {
      WAIT_ASYNC(0);
    }
    __syncthreads();

    const _Float16* Kc = KstB + cur * (32 * 256);
    const _Float16* Vc = VstB + cur * (256 * 32);

    v8f s0 = {}, s1 = {};
#pragma unroll
    for (int kb = 0; kb < 8; ++kb) {
      v16h k0 = load_b16(Kc + lid * 256 + kb * 32, half_);
      v16h k1 = load_b16(Kc + (16 + lid) * 256 + kb * 32, half_);
      s0 = wmma_f16(qa[kb], k0, s0);
      s1 = wmma_f16(qa[kb], k1, s1);
    }

    float alpha[8];
#pragma unroll
    for (int r = 0; r < 8; ++r) {
      float mx = fmaxf(s0[r], s1[r]);
#pragma unroll
      for (int msk = 8; msk >= 1; msk >>= 1)
        mx = fmaxf(mx, __shfl_xor(mx, msk, 32));
      const float mnew = fmaxf(mrun[r], mx);
      const float a    = __expf(mrun[r] - mnew);
      const float p0   = __expf(s0[r] - mnew);
      const float p1   = __expf(s1[r] - mnew);
      s0[r] = p0; s1[r] = p1;
      float rs = p0 + p1;
#pragma unroll
      for (int msk = 8; msk >= 1; msk >>= 1) rs += __shfl_xor(rs, msk, 32);
      lrun[r]  = lrun[r] * a + rs;
      mrun[r]  = mnew;
      alpha[r] = a;
    }

#pragma unroll
    for (int t = 0; t < 16; ++t)
#pragma unroll
      for (int r = 0; r < 8; ++r) out[t][r] *= alpha[r];

    // P: C/D layout -> A-operand layout via wave-private LDS (in-order DS).
    _Float16* pw = pb + wv * (16 * 32);
#pragma unroll
    for (int r = 0; r < 8; ++r) {
      pw[(r + half_ * 8) * 32 + lid]      = (_Float16)s0[r];
      pw[(r + half_ * 8) * 32 + 16 + lid] = (_Float16)s1[r];
    }
    v16h pa = load_a16(pw + lid * 32, half_);

#pragma unroll
    for (int t = 0; t < 16; ++t) {
      v16h vb = load_b16(Vc + (t * 16 + lid) * 32, half_);
      out[t] = wmma_f16(pa, vb, out[t]);
    }
    __syncthreads();                       // done reading buf before restage
  }

  float inv[8];
#pragma unroll
  for (int r = 0; r < 8; ++r) inv[r] = 1.0f / lrun[r];
#pragma unroll
  for (int t = 0; t < 16; ++t) {
    const int d = t * 16 + lid;
#pragma unroll
    for (int r = 0; r < 8; ++r) {
      const int row = q0 + r + half_ * 8;
      AO[((size_t)b * N_ + row) * D_ + d] = (_Float16)(out[t][r] * inv[r]);
    }
  }
}

// ---------------------------------------------------------------------------
// Kernel 3: output projection + bias + residual -> Y f32 [B,C,N].
__global__ void __launch_bounds__(256) out_proj_kernel(
    const _Float16* __restrict__ AO, const _Float16* __restrict__ Woh,
    const float* __restrict__ bo, const float* __restrict__ x,
    float* __restrict__ Y) {
  const int wave  = (blockIdx.x * blockDim.x + threadIdx.x) >> 5;
  const int ln    = threadIdx.x & 31;
  const int half_ = ln >> 4;
  const int lid   = ln & 15;
  const int b     = wave / (N_ / 16);
  const int q0    = (wave % (N_ / 16)) * 16;

  v16h aa[8];
  {
    const _Float16* rowp = AO + ((size_t)b * N_ + q0 + lid) * D_;
#pragma unroll
    for (int kb = 0; kb < 8; ++kb) aa[kb] = load_a16(rowp + kb * 32, half_);
  }

  for (int ct = 0; ct < 16; ++ct) {
    const int c = ct * 16 + lid;
    const _Float16* wrow = Woh + (size_t)c * D_;
    v8f acc = {};
#pragma unroll
    for (int kb = 0; kb < 8; ++kb)
      acc = wmma_f16(aa[kb], load_b16(wrow + kb * 32, half_), acc);
    const float bc = bo[c];
#pragma unroll
    for (int r = 0; r < 8; ++r) {
      const int n = q0 + r + half_ * 8;
      const size_t idx = ((size_t)b * C_ + c) * N_ + n;
      Y[idx] = acc[r] + bc + x[idx];
    }
  }
}

// ---------------------------------------------------------------------------
// Kernel 4: GroupNorm stats; group = 8 contiguous channels = 32768 floats.
__global__ void __launch_bounds__(256) gn_stats_kernel(
    const float* __restrict__ Y, float* __restrict__ stats) {
  const int b = blockIdx.x / NG;
  const int g = blockIdx.x % NG;
  const float* p = Y + ((size_t)b * C_ + (size_t)g * CPG) * N_;
  float s = 0.0f, ss = 0.0f;
  for (int i = threadIdx.x; i < CPG * N_; i += 256) {
    const float v = p[i];
    s += v; ss += v * v;
  }
#pragma unroll
  for (int msk = 16; msk >= 1; msk >>= 1) {
    s  += __shfl_xor(s, msk, 32);
    ss += __shfl_xor(ss, msk, 32);
  }
  __shared__ float shs[8], shss[8];
  const int wv = threadIdx.x >> 5;
  if ((threadIdx.x & 31) == 0) { shs[wv] = s; shss[wv] = ss; }
  __syncthreads();
  if (threadIdx.x == 0) {
    float S = 0.0f, SS = 0.0f;
#pragma unroll
    for (int i = 0; i < 8; ++i) { S += shs[i]; SS += shss[i]; }
    const float cnt  = (float)(CPG * N_);
    const float mean = S / cnt;
    const float var  = SS / cnt - mean * mean;
    stats[blockIdx.x * 2 + 0] = mean;
    stats[blockIdx.x * 2 + 1] = rsqrtf(var + GN_EPS);
  }
}

// Kernel 5: GroupNorm apply (in place on d_out).
__global__ void __launch_bounds__(256) gn_apply_kernel(
    float* __restrict__ Y, const float* __restrict__ stats,
    const float* __restrict__ gamma, const float* __restrict__ beta) {
  const size_t i = (size_t)blockIdx.x * 256 + threadIdx.x;
  const int c = (int)((i / N_) % C_);
  const int b = (int)(i / ((size_t)C_ * N_));
  const int gidx = b * NG + c / CPG;
  Y[i] = (Y[i] - stats[gidx * 2]) * stats[gidx * 2 + 1] * gamma[c] + beta[c];
}

// ---------------------------------------------------------------------------
extern "C" void kernel_launch(void* const* d_in, const int* in_sizes, int n_in,
                              void* d_out, int out_size, void* d_ws, size_t ws_size,
                              hipStream_t stream) {
  const float* x     = (const float*)d_in[0];
  const float* Wq    = (const float*)d_in[1];
  const float* bq    = (const float*)d_in[2];
  const float* Wk    = (const float*)d_in[3];
  const float* bk    = (const float*)d_in[4];
  const float* Wv    = (const float*)d_in[5];
  const float* bv    = (const float*)d_in[6];
  const float* Wo    = (const float*)d_in[7];
  const float* bo    = (const float*)d_in[8];
  const float* gamma = (const float*)d_in[9];
  const float* beta  = (const float*)d_in[10];

  char* ws = (char*)d_ws;
  const size_t seg  = (size_t)B_ * N_ * D_ * sizeof(_Float16);  // 8 MB
  const size_t wseg = (size_t)D_ * C_ * sizeof(_Float16);       // 128 KB
  _Float16* Qh  = (_Float16*)(ws);
  _Float16* Kh  = (_Float16*)(ws + seg);
  _Float16* Vh  = (_Float16*)(ws + 2 * seg);
  _Float16* AO  = (_Float16*)(ws + 3 * seg);
  _Float16* Xh  = (_Float16*)(ws + 4 * seg);          // [B,N,C] f16, 8 MB
  _Float16* Wqh = (_Float16*)(ws + 5 * seg);
  _Float16* Wkh = (_Float16*)(ws + 5 * seg + wseg);
  _Float16* Wvh = (_Float16*)(ws + 5 * seg + 2 * wseg);
  _Float16* Woh = (_Float16*)(ws + 5 * seg + 3 * wseg);
  float* stats  = (float*)(ws + 5 * seg + 4 * wseg);
  float* Y      = (float*)d_out;                       // reuse out as scratch

  const int blocks = (B_ * (N_ / 16)) / 8;             // 128
  const int wn = D_ * C_;

  wcvt_kernel<<<wn / 256, 256, 0, stream>>>(Wq, Wqh, wn);
  wcvt_kernel<<<wn / 256, 256, 0, stream>>>(Wk, Wkh, wn);
  wcvt_kernel<<<wn / 256, 256, 0, stream>>>(Wv, Wvh, wn);
  wcvt_kernel<<<wn / 256, 256, 0, stream>>>(Wo, Woh, wn);
  {
    dim3 g(B_ * C_ / 32, N_ / 32);
    xpose_kernel<<<g, 256, 0, stream>>>(x, Xh);
  }
  qkv_proj_kernel<<<blocks, 256, 0, stream>>>(Xh, Wqh, bq, Wkh, bk, Wvh, bv,
                                              Qh, Kh, Vh);
  const size_t flash_lds = (size_t)(KST_H + VST_H + 8 * 16 * 32) * sizeof(_Float16);
  flash_attn_kernel<<<blocks, 256, flash_lds, stream>>>(Qh, Kh, Vh, AO);
  out_proj_kernel<<<blocks, 256, 0, stream>>>(AO, Woh, bo, x, Y);
  gn_stats_kernel<<<B_ * NG, 256, 0, stream>>>(Y, stats);
  gn_apply_kernel<<<(B_ * C_ * N_) / 256, 256, 0, stream>>>(Y, stats, gamma, beta);
}
